// StructureEvaluator_36455682408741
// MI455X (gfx1250) — matrix operand order repaired and checked by reading, not compiled, not run on hardware
//
#include <hip/hip_runtime.h>
#include <hip/hip_bf16.h>

typedef __attribute__((ext_vector_type(16))) _Float16 v16h;
typedef __attribute__((ext_vector_type(8)))  _Float16 v8h;
typedef __attribute__((ext_vector_type(8)))  float    v8f;

#define DEVINL __device__ __forceinline__

// ---------------------------------------------------------------------------
// WMMA fragment loaders (v_wmma_f32_16x16x32_f16 layouts, wave32)
// A: 16x32 f16, row-major source with leading dim lda.
//   lane L: row = L&15; khalf = L>>4; a[0..7] = K[k0+8*khalf ..], a[8..15] = +16
// B: 32x16 (K x N) sourced from W stored (N x K) row-major (PyTorch Linear).
//   lane L: col = L&15; khalf = L>>4; b[0..7] = W[col, k0+8*khalf ..], b[8..15] = +16
// C/D: v8f; lane L: col = L&15; rows = (L>>4)*8 + j
// ---------------------------------------------------------------------------
DEVINL v16h frag_a(const _Float16* A, int lda, int row0, int k0, int lane) {
    int r  = row0 + (lane & 15);
    int kh = lane >> 4;
    const _Float16* p = A + (size_t)r * lda + k0 + kh * 8;
    v8h lo = *(const v8h*)p;
    v8h hi = *(const v8h*)(p + 16);
    v16h a;
#pragma unroll
    for (int i = 0; i < 8; ++i) { a[i] = lo[i]; a[8 + i] = hi[i]; }
    return a;
}

DEVINL v16h frag_b(const _Float16* W, int ldw, int col0, int k0, int lane) {
    int c  = col0 + (lane & 15);
    int kh = lane >> 4;
    const _Float16* p = W + (size_t)c * ldw + k0 + kh * 8;
    v8h lo = *(const v8h*)p;
    v8h hi = *(const v8h*)(p + 16);
    v16h b;
#pragma unroll
    for (int i = 0; i < 8; ++i) { b[i] = lo[i]; b[8 + i] = hi[i]; }
    return b;
}

DEVINL v8f wmma32(v16h a, v16h b, v8f c) {
    return __builtin_amdgcn_wmma_f32_16x16x32_f16(false, a, false, b,
                                                  (short)0, c, false, false);
}

// ---------------------------------------------------------------------------
// GEMM: C[M,Nc] = A[M,K] * W[Nc,K]^T (+bias). Each wave: 16 rows x 64 cols
// (4 accumulators), sharing one A fragment per k-step across 4 WMMAs.
// Nc must be a multiple of 64.
// ---------------------------------------------------------------------------
__global__ void k_gemm_wmma(const _Float16* __restrict__ A,
                            const _Float16* __restrict__ W,
                            const float* __restrict__ bias,
                            float* __restrict__ C,
                            _Float16* __restrict__ Ch,
                            int M, int Nc, int K) {
    int wave = threadIdx.x >> 5;
    int lane = threadIdx.x & 31;
    int jobs_n = Nc >> 6;                       // 64-wide column panels
    int total  = (M >> 4) * jobs_n;
    int job    = blockIdx.x * 8 + wave;
    if (job >= total) return;                   // wave-uniform exit
    int jm = job / jobs_n, jn = job - jm * jobs_n;
    int row0 = jm << 4, col0 = jn << 6;

    // prefetch next row panel of A (speculative; dropped if OOB)
    if (row0 + 16 < M)
        __builtin_prefetch(A + (size_t)(row0 + 16) * K, 0, 0);

    v8f acc0 = {0.f,0.f,0.f,0.f,0.f,0.f,0.f,0.f};
    v8f acc1 = acc0, acc2 = acc0, acc3 = acc0;
    for (int k0 = 0; k0 < K; k0 += 32) {
        v16h a  = frag_a(A, K, row0, k0, lane);
        v16h b0 = frag_b(W, K, col0,      k0, lane);
        v16h b1 = frag_b(W, K, col0 + 16, k0, lane);
        v16h b2 = frag_b(W, K, col0 + 32, k0, lane);
        v16h b3 = frag_b(W, K, col0 + 48, k0, lane);
        acc0 = wmma32(a, b0, acc0);
        acc1 = wmma32(a, b1, acc1);
        acc2 = wmma32(a, b2, acc2);
        acc3 = wmma32(a, b3, acc3);
    }
    int r0 = row0 + ((lane >> 4) << 3);
    v8f accs[4] = {acc0, acc1, acc2, acc3};
#pragma unroll
    for (int s = 0; s < 4; ++s) {
        int cn = col0 + s * 16 + (lane & 15);
        float bv = bias ? bias[cn] : 0.f;
#pragma unroll
        for (int j = 0; j < 8; ++j) {
            float v = accs[s][j] + bv;
            size_t idx = (size_t)(r0 + j) * Nc + cn;
            if (C)  C[idx]  = v;
            if (Ch) Ch[idx] = (_Float16)v;
        }
    }
}

// ---------------------------------------------------------------------------
__global__ void k_f32_to_f16(const float* __restrict__ in,
                             _Float16* __restrict__ out, int n) {
    int i = blockIdx.x * blockDim.x + threadIdx.x;
    if (i < n) out[i] = (_Float16)in[i];
}

// emb[b,n,:] = table[x[b,n],:] + pos[n,:]   -> f16
__global__ void k_embed(const int* __restrict__ x, const float* __restrict__ tab,
                        const float* __restrict__ pos, _Float16* __restrict__ out) {
    int idx = blockIdx.x * blockDim.x + threadIdx.x;   // B*N*128 = 2097152
    if (idx >= 32 * 512 * 128) return;
    int c  = idx & 127;
    int bn = idx >> 7;            // b*512+n
    int n  = bn & 511;
    int tok = x[bn];
    float v = tab[tok * 128 + c] + pos[n * 128 + c];
    out[idx] = (_Float16)v;
}

// f/g attention scores: one wave per (node, head); F = 128 fixed.
__global__ void k_fg(const _Float16* __restrict__ Wh, const float* __restrict__ avec,
                     float* __restrict__ f, float* __restrict__ g,
                     int heads, int rowStride) {
    int lane = threadIdx.x & 31;
    int w = blockIdx.x * (blockDim.x >> 5) + (threadIdx.x >> 5);
    int total = 32 * 512 * heads;
    if (w >= total) return;
    int h = w % heads;
    int node = w / heads;
    const _Float16* row = Wh + (size_t)node * rowStride + h * 128;
    const float* a1 = avec + h * 256;
    const float* a2 = a1 + 128;
    float sf = 0.f, sg = 0.f;
#pragma unroll
    for (int q = 0; q < 4; ++q) {
        int k = lane + q * 32;
        float v = (float)row[k];
        sf += v * a1[k];
        sg += v * a2[k];
    }
    for (int off = 16; off; off >>= 1) {
        sf += __shfl_down(sf, off);
        sg += __shfl_down(sg, off);
    }
    if (lane == 0) { f[w] = sf; g[w] = sg; }
}

// Banded (|i-j|<=3) softmax attention + weighted sum. One wave per (b,i,head).
__global__ void k_attn(const _Float16* __restrict__ Wh,
                       const float* __restrict__ f, const float* __restrict__ g,
                       _Float16* __restrict__ outH, float* __restrict__ outF,
                       int heads, int rowStride, int doElu) {
    int lane = threadIdx.x & 31;
    int w = blockIdx.x * (blockDim.x >> 5) + (threadIdx.x >> 5);
    int total = 32 * 512 * heads;
    if (w >= total) return;
    int h = w % heads;
    int node = w / heads;
    int b = node >> 9;
    int i = node & 511;
    float fi = f[w];
    int jlo = i - 3; if (jlo < 0) jlo = 0;
    int jhi = i + 3; if (jhi > 511) jhi = 511;
    int cnt = jhi - jlo + 1;
    float wj[7];
    float m = -3.4e38f;
    for (int t = 0; t < cnt; ++t) {
        int j = jlo + t;
        float e = fi + g[(((size_t)b << 9) + j) * heads + h];
        e = e > 0.f ? e : 0.2f * e;          // LeakyReLU(0.2)
        wj[t] = e;
        if (e > m) m = e;
    }
    float s = 0.f;
    for (int t = 0; t < cnt; ++t) { wj[t] = expf(wj[t] - m); s += wj[t]; }
    float inv = 1.f / s;
    for (int t = 0; t < cnt; ++t) wj[t] *= inv;
#pragma unroll
    for (int q = 0; q < 4; ++q) {
        int c = lane + q * 32;
        float acc = 0.f;
        for (int t = 0; t < cnt; ++t) {
            int j = jlo + t;
            acc += wj[t] * (float)Wh[((size_t)b * 512 + j) * rowStride + h * 128 + c];
        }
        if (doElu) acc = acc > 0.f ? acc : (expf(acc) - 1.f);   // ELU
        size_t o = (size_t)node * rowStride + h * 128 + c;
        if (outH) outH[o] = (_Float16)acc;
        if (outF) outF[o] = acc;
    }
}

// Mean-pool over N: one wave per (b, feature).
__global__ void k_pool(const float* __restrict__ gatOut, float* __restrict__ pooled) {
    int lane = threadIdx.x & 31;
    int w = blockIdx.x * (blockDim.x >> 5) + (threadIdx.x >> 5);
    if (w >= 32 * 128) return;
    int b = w >> 7, c = w & 127;
    float s = 0.f;
    for (int n = lane; n < 512; n += 32)
        s += gatOut[((size_t)b * 512 + n) * 128 + c];
    for (int off = 16; off; off >>= 1) s += __shfl_down(s, off);
    if (lane == 0) pooled[w] = s * (1.f / 512.f);
}

// ---------------------------------------------------------------------------
// GRU recurrence: blockIdx.x = direction (0 fwd, 1 bwd), 256 threads = 8 waves.
// Whh staged in LDS once (96KB f16); per step gh(32x384) = h(32x128) @ Whh^T
// via WMMA. Wave w owns row-panel tm = w&1 and 6 column tiles, so its 4
// A-fragments are reused across 24 WMMAs per step. h kept in LDS (f32 + f16).
// Dynamic LDS: Whh 98304 + gh 49152 + h32 16384 + h16 8192 = 172032 B.
// ---------------------------------------------------------------------------
__global__ void k_gru(const float* __restrict__ giF, const float* __restrict__ giB,
                      const _Float16* __restrict__ WhhF, const _Float16* __restrict__ WhhB,
                      const float* __restrict__ bhhF, const float* __restrict__ bhhB,
                      float* __restrict__ hOutF, float* __restrict__ hOutB) {
    extern __shared__ char smem[];
    _Float16* whhS = (_Float16*)smem;                               // 384*128 f16
    float*    gh   = (float*)(smem + 98304);                        // 32*384
    float*    h32  = (float*)(smem + 98304 + 49152);                // 32*128
    _Float16* h16  = (_Float16*)(smem + 98304 + 49152 + 16384);     // 32*128

    const int fwd = (blockIdx.x == 0);
    const float*    gi   = fwd ? giF  : giB;
    const _Float16* Whh  = fwd ? WhhF : WhhB;
    const float*    bhh  = fwd ? bhhF : bhhB;
    float*          hOut = fwd ? hOutF : hOutB;

    int tid  = threadIdx.x;
    int lane = tid & 31, wave = tid >> 5;

    // stage Whh into LDS (49152 halves = 24576 dwords), init h = 0
    {
        const unsigned int* src = (const unsigned int*)Whh;
        unsigned int* dst = (unsigned int*)whhS;
        for (int i = tid; i < 384 * 128 / 2; i += 256) dst[i] = src[i];
    }
    for (int e = tid; e < 32 * 128; e += 256) { h32[e] = 0.f; h16[e] = (_Float16)0.f; }
    __syncthreads();

    const int tm   = wave & 1;          // row panel (batch rows 0-15 / 16-31)
    const int tnb  = (wave >> 1) * 6;   // first of 6 column tiles
    const int row0 = tm << 4;

    for (int t = 0; t < 512; ++t) {
        // A fragments for this wave's row panel (from LDS h16)
        v16h a0 = frag_a((const _Float16*)h16, 128, row0,  0, lane);
        v16h a1 = frag_a((const _Float16*)h16, 128, row0, 32, lane);
        v16h a2 = frag_a((const _Float16*)h16, 128, row0, 64, lane);
        v16h a3 = frag_a((const _Float16*)h16, 128, row0, 96, lane);
#pragma unroll
        for (int q = 0; q < 6; ++q) {
            int col0 = (tnb + q) << 4;
            v8f acc = {0.f,0.f,0.f,0.f,0.f,0.f,0.f,0.f};
            acc = wmma32(a0, frag_b((const _Float16*)whhS, 128, col0,  0, lane), acc);
            acc = wmma32(a1, frag_b((const _Float16*)whhS, 128, col0, 32, lane), acc);
            acc = wmma32(a2, frag_b((const _Float16*)whhS, 128, col0, 64, lane), acc);
            acc = wmma32(a3, frag_b((const _Float16*)whhS, 128, col0, 96, lane), acc);
            int cn = col0 + (lane & 15);
            int r0 = row0 + ((lane >> 4) << 3);
#pragma unroll
            for (int j = 0; j < 8; ++j) gh[(r0 + j) * 384 + cn] = acc[j];
        }
        __syncthreads();

        int n = fwd ? t : (511 - t);
        for (int e = tid; e < 32 * 128; e += 256) {
            int b = e >> 7, j = e & 127;
            const float* gir = gi + ((size_t)b * 512 + n) * 384;   // bih already fused
            float ir = gir[j], iz = gir[128 + j], inn = gir[256 + j];
            float hr = gh[b * 384 + j]       + bhh[j];
            float hz = gh[b * 384 + 128 + j] + bhh[128 + j];
            float hn = gh[b * 384 + 256 + j] + bhh[256 + j];
            float r  = 1.f / (1.f + expf(-(ir + hr)));
            float z  = 1.f / (1.f + expf(-(iz + hz)));
            float nn = tanhf(inn + r * hn);
            float hp = h32[e];
            float hv = (1.f - z) * nn + z * hp;
            h32[e] = hv;
            h16[e] = (_Float16)hv;
        }
        __syncthreads();
    }
    for (int e = tid; e < 32 * 128; e += 256) hOut[e] = h32[e];
}

// ---------------------------------------------------------------------------
// Output heads: one block (256 threads) per batch element.
// ---------------------------------------------------------------------------
__global__ void k_heads(const float* __restrict__ pooled,
                        const float* __restrict__ hF, const float* __restrict__ hB,
                        const float* __restrict__ s1W, const float* __restrict__ s1b,
                        const float* __restrict__ lng, const float* __restrict__ lnb,
                        const float* __restrict__ s2W, const float* __restrict__ s2b,
                        const float* __restrict__ s3W, const float* __restrict__ s3b,
                        const float* __restrict__ ss1W, const float* __restrict__ ss1b,
                        const float* __restrict__ ss2W, const float* __restrict__ ss2b,
                        float* __restrict__ out) {
    __shared__ float comb[384];
    __shared__ float hbuf[256];
    __shared__ float h2buf[128];
    __shared__ float hsbuf[256];
    __shared__ float red[256];
    int b = blockIdx.x, tid = threadIdx.x;
    if (tid < 128) {
        comb[tid]       = pooled[b * 128 + tid];
        comb[128 + tid] = hF[b * 128 + tid];
        comb[256 + tid] = hB[b * 128 + tid];
    }
    __syncthreads();
    // stability: s1 -> LN -> relu
    float hv;
    {
        const float* wr = s1W + (size_t)tid * 384;
        float s = 0.f;
        for (int k = 0; k < 384; ++k) s += comb[k] * wr[k];
        hv = s + s1b[tid];
    }
    red[tid] = hv; __syncthreads();
    for (int o = 128; o; o >>= 1) { if (tid < o) red[tid] += red[tid + o]; __syncthreads(); }
    float mu = red[0] / 256.f; __syncthreads();
    float d = hv - mu;
    red[tid] = d * d; __syncthreads();
    for (int o = 128; o; o >>= 1) { if (tid < o) red[tid] += red[tid + o]; __syncthreads(); }
    float var = red[0] / 256.f; __syncthreads();
    float ln = d * rsqrtf(var + 1e-5f) * lng[tid] + lnb[tid];
    hbuf[tid] = ln > 0.f ? ln : 0.f;
    // ss head layer 1 (also consumes comb)
    {
        const float* wr = ss1W + (size_t)tid * 384;
        float s = 0.f;
        for (int k = 0; k < 384; ++k) s += comb[k] * wr[k];
        s += ss1b[tid];
        hsbuf[tid] = s > 0.f ? s : 0.f;
    }
    __syncthreads();
    if (tid < 128) {
        const float* wr = s2W + (size_t)tid * 256;
        float s = 0.f;
        for (int k = 0; k < 256; ++k) s += hbuf[k] * wr[k];
        s += s2b[tid];
        h2buf[tid] = s > 0.f ? s : 0.f;
    }
    __syncthreads();
    red[tid] = (tid < 128) ? h2buf[tid] * s3W[tid] : 0.f;
    __syncthreads();
    for (int o = 128; o; o >>= 1) { if (tid < o) red[tid] += red[tid + o]; __syncthreads(); }
    if (tid == 0) {
        float v = red[0] + s3b[0];
        out[b] = 1.f / (1.f + expf(-v));
    }
    if (tid < 3) {
        const float* wr = ss2W + (size_t)tid * 256;
        float s = 0.f;
        for (int k = 0; k < 256; ++k) s += hsbuf[k] * wr[k];
        out[32 + b * 3 + tid] = s + ss2b[tid];
    }
}

// ---------------------------------------------------------------------------
extern "C" void kernel_launch(void* const* d_in, const int* in_sizes, int n_in,
                              void* d_out, int out_size, void* d_ws, size_t ws_size,
                              hipStream_t stream) {
    (void)in_sizes; (void)n_in; (void)out_size; (void)ws_size;
    const int*   x     = (const int*)  d_in[0];
    const float* embT  = (const float*)d_in[1];
    const float* pos   = (const float*)d_in[2];
    const float* gat1W = (const float*)d_in[3];
    const float* gat1a = (const float*)d_in[4];
    const float* gat2W = (const float*)d_in[5];
    const float* gat2a = (const float*)d_in[6];
    const float* WihF  = (const float*)d_in[7];
    const float* WhhF  = (const float*)d_in[8];
    const float* bihF  = (const float*)d_in[9];
    const float* bhhF  = (const float*)d_in[10];
    const float* WihB  = (const float*)d_in[11];
    const float* WhhB  = (const float*)d_in[12];
    const float* bihB  = (const float*)d_in[13];
    const float* bhhB  = (const float*)d_in[14];
    const float* s1W   = (const float*)d_in[15];
    const float* s1b   = (const float*)d_in[16];
    const float* lng   = (const float*)d_in[17];
    const float* lnb   = (const float*)d_in[18];
    const float* s2W   = (const float*)d_in[19];
    const float* s2b   = (const float*)d_in[20];
    const float* s3W   = (const float*)d_in[21];
    const float* s3b   = (const float*)d_in[22];
    const float* ss1W  = (const float*)d_in[23];
    const float* ss1b  = (const float*)d_in[24];
    const float* ss2W  = (const float*)d_in[25];
    const float* ss2b  = (const float*)d_in[26];
    float* out = (float*)d_out;

    char* ws = (char*)d_ws;
    size_t off = 0;
    auto carve = [&](size_t bytes) -> char* {
        char* p = ws + off;
        off = (off + bytes + 255) & ~(size_t)255;
        return p;
    };
    _Float16* emb16  = (_Float16*)carve(32 * 512 * 128 * 2);   // B,N,EMB
    _Float16* w1h    = (_Float16*)carve(512 * 128 * 2);        // gat1_W f16 (512x128)
    _Float16* w2h    = (_Float16*)carve(128 * 512 * 2);        // gat2_W f16
    _Float16* wihF16 = (_Float16*)carve(384 * 128 * 2);
    _Float16* whhF16 = (_Float16*)carve(384 * 128 * 2);
    _Float16* wihB16 = (_Float16*)carve(384 * 128 * 2);
    _Float16* whhB16 = (_Float16*)carve(384 * 128 * 2);
    _Float16* wh1h   = (_Float16*)carve((size_t)32 * 512 * 512 * 2); // GAT1 proj (f16)
    float*    f1     = (float*)   carve((size_t)32 * 512 * 4 * 4);
    float*    g1     = (float*)   carve((size_t)32 * 512 * 4 * 4);
    _Float16* h1h    = (_Float16*)carve((size_t)32 * 512 * 512 * 2); // GAT1 out (f16)
    _Float16* wh2h   = (_Float16*)carve((size_t)32 * 512 * 128 * 2); // GAT2 proj (f16)
    float*    f2     = (float*)   carve((size_t)32 * 512 * 4);
    float*    g2     = (float*)   carve((size_t)32 * 512 * 4);
    float*    out2   = (float*)   carve((size_t)32 * 512 * 128 * 4); // GAT2 out (f32)
    float*    pooled = (float*)   carve(32 * 128 * 4);
    float*    giF    = (float*)   carve((size_t)32 * 512 * 384 * 4);
    float*    giB    = (float*)   carve((size_t)32 * 512 * 384 * 4);
    float*    hF     = (float*)   carve(32 * 128 * 4);
    float*    hB     = (float*)   carve(32 * 128 * 4);

    // weight conversions to f16
    k_f32_to_f16<<<(65536 + 255) / 256, 256, 0, stream>>>(gat1W, w1h, 65536);
    k_f32_to_f16<<<(65536 + 255) / 256, 256, 0, stream>>>(gat2W, w2h, 65536);
    k_f32_to_f16<<<(49152 + 255) / 256, 256, 0, stream>>>(WihF, wihF16, 49152);
    k_f32_to_f16<<<(49152 + 255) / 256, 256, 0, stream>>>(WhhF, whhF16, 49152);
    k_f32_to_f16<<<(49152 + 255) / 256, 256, 0, stream>>>(WihB, wihB16, 49152);
    k_f32_to_f16<<<(49152 + 255) / 256, 256, 0, stream>>>(WhhB, whhB16, 49152);

    // embedding + positional
    k_embed<<<8192, 256, 0, stream>>>(x, embT, pos, emb16);

    // GAT1 projection: (16384x128) x (512x128)^T -> wh1h   [8192 wave-jobs]
    k_gemm_wmma<<<1024, 256, 0, stream>>>(emb16, w1h, nullptr, nullptr, wh1h,
                                          16384, 512, 128);
    k_fg<<<8192, 256, 0, stream>>>(wh1h, gat1a, f1, g1, 4, 512);
    k_attn<<<8192, 256, 0, stream>>>(wh1h, f1, g1, h1h, nullptr, 4, 512, 1);

    // GAT2 projection: (16384x512) x (128x512)^T -> wh2h   [2048 wave-jobs]
    k_gemm_wmma<<<256, 256, 0, stream>>>(h1h, w2h, nullptr, nullptr, wh2h,
                                         16384, 128, 512);
    k_fg<<<2048, 256, 0, stream>>>(wh2h, gat2a, f2, g2, 1, 128);
    k_attn<<<2048, 256, 0, stream>>>(wh2h, f2, g2, nullptr, out2, 1, 128, 0);
    k_pool<<<512, 256, 0, stream>>>(out2, pooled);

    // GRU input projections (bih fused): (16384x128) x (384x128)^T  [6144 jobs]
    k_gemm_wmma<<<768, 256, 0, stream>>>(emb16, wihF16, bihF, giF, nullptr,
                                         16384, 384, 128);
    k_gemm_wmma<<<768, 256, 0, stream>>>(emb16, wihB16, bihB, giB, nullptr,
                                         16384, 384, 128);

    // GRU recurrence (one WGP per direction), 172032 B dynamic LDS
    k_gru<<<2, 256, 172032, stream>>>(giF, giB, whhF16, whhB16, bhhF, bhhB, hF, hB);

    // heads
    k_heads<<<32, 256, 0, stream>>>(pooled, hF, hB, s1W, s1b, lng, lnb,
                                    s2W, s2b, s3W, s3b, ss1W, ss1b, ss2W, ss2b, out);
}
